// NonlocalSingleBlock_44186623541795
// MI455X (gfx1250) — matrix operand order repaired and checked by reading, not compiled, not run on hardware
//
#include <hip/hip_runtime.h>
#include <hip/hip_bf16.h>

#define B_ 8
#define C_ 256
#define HC_ 128
#define N_ 2048
#define SLD (N_ + 8)   // bf16 score-strip row stride (keeps 16B align + bank spread)

typedef __attribute__((ext_vector_type(16))) __bf16 v16bf;
typedef __attribute__((ext_vector_type(8)))  __bf16 v8bf;
typedef __attribute__((ext_vector_type(8)))  float  v8f;
typedef __attribute__((ext_vector_type(4)))  unsigned int v4u;
typedef __attribute__((ext_vector_type(8)))  int v8i;
typedef __attribute__((ext_vector_type(4)))  int v4i;

// ---------------------------------------------------------------------------
// WMMA fragment helpers (CDNA5 wave32, v_wmma_f32_16x16x32_bf16)
// A-frag 16(M)x32(K): lane<16: row=lane, k=[0..7]+[16..23]; lane>=16: k=[8..15]+[24..31]
// B-frag 32(K)x16(N): source column-major [N][K]; lane<16: col=lane, k=[0..7]+[16..23];
//                     lane>=16: col=lane-16, k=[8..15]+[24..31]
// ---------------------------------------------------------------------------
__device__ inline v16bf load_a_frag(const __bf16* base, int ld) {
    int lane = threadIdx.x & 31;
    int row  = lane & 15;
    int kb   = (lane >> 4) << 3;
    const __bf16* p = base + (size_t)row * ld + kb;
    union { v16bf v; v8bf h[2]; } u;
    u.h[0] = *(const v8bf*)(p);
    u.h[1] = *(const v8bf*)(p + 16);
    return u.v;
}

__device__ inline v16bf load_b_frag(const __bf16* base, int ld) {
    int lane = threadIdx.x & 31;
    int col  = lane & 15;
    int kb   = (lane >> 4) << 3;
    const __bf16* p = base + (size_t)col * ld + kb;
    union { v16bf v; v8bf h[2]; } u;
    u.h[0] = *(const v8bf*)(p);
    u.h[1] = *(const v8bf*)(p + 16);
    return u.v;
}

__device__ inline v8f wmma_bf16(v16bf a, v16bf b, v8f c) {
    return __builtin_amdgcn_wmma_f32_16x16x32_bf16(
        false, a, false, b, (short)0, c, false, false);
}

// ---------------------------------------------------------------------------
// TDM: 2D tile load Global -> LDS (D# per CDNA5 ISA ch.8; data_size=2B, type=2).
// Arity differs per toolchain (ROCm7.2/clang-22: 5 args, clang-23: 6 args).
// ---------------------------------------------------------------------------
__device__ inline void tdm_load_2d_bf16(const __bf16* gsrc, __bf16* lds_dst,
                                        unsigned width, unsigned height,
                                        unsigned row_stride) {
    unsigned long long ga = (unsigned long long)(size_t)gsrc;
    unsigned lds_off = (unsigned)(size_t)lds_dst;       // LDS aperture: offset = addr[31:0]
    v4u g0 = {0u, 0u, 0u, 0u};
    g0[0] = 1u;                                          // count=1, user descriptor
    g0[1] = lds_off;                                     // lds_addr (bytes)
    g0[2] = (unsigned)(ga & 0xffffffffULL);              // global_addr[31:0]
    g0[3] = (unsigned)((ga >> 32) & 0x01ffffffULL) | (2u << 30);   // addr[56:32], type=2
    v8i g1 = {0, 0, 0, 0, 0, 0, 0, 0};
    g1[0] = (1 << 16);                                   // workgroup_mask=0, data_size=1 (2B)
    g1[1] = (int)((width & 0xffffu) << 16);              // tensor_dim0[15:0]
    g1[2] = (int)(((width >> 16) & 0xffffu) | ((height & 0xffffu) << 16)); // dim0 hi / dim1 lo
    g1[3] = (int)(((height >> 16) & 0xffffu) | ((width & 0xffffu) << 16)); // dim1 hi / tile_dim0
    g1[4] = (int)(height & 0xffffu);                     // tile_dim1 (tile_dim2=0)
    g1[5] = (int)row_stride;                             // tensor_dim0_stride[31:0]
    v4i gz = {0, 0, 0, 0};
#if defined(__clang_major__) && __clang_major__ >= 23
    v8i gz8 = {0, 0, 0, 0, 0, 0, 0, 0};
    __builtin_amdgcn_tensor_load_to_lds(g0, g1, gz, gz, gz8, 0);
#else
    __builtin_amdgcn_tensor_load_to_lds(g0, g1, gz, gz, 0);
#endif
}

// ---------------------------------------------------------------------------
// Small prep kernels
// ---------------------------------------------------------------------------
__global__ void cvt_f32_bf16(const float* __restrict__ src, __bf16* __restrict__ dst, int n) {
    int i = blockIdx.x * 256 + threadIdx.x;
    if (i < n) dst[i] = (__bf16)src[i];
}

__global__ void bn_prep(const float* g1, const float* be1, const float* m1, const float* v1,
                        const float* g2, const float* be2, const float* m2, const float* v2,
                        float* inv1, float* sh1, float* inv2, float* sh2) {
    int t = threadIdx.x;
    if (t < HC_) {
        float iv = g1[t] * rsqrtf(v1[t] + 1e-5f);
        inv1[t] = iv; sh1[t] = be1[t] - m1[t] * iv;
    } else {
        int u = t - HC_;
        float iv = g2[u] * rsqrtf(v2[u] + 1e-5f);
        inv2[u] = iv; sh2[u] = be2[u] - m2[u] * iv;
    }
}

// x [B][C][N] f32  ->  xt [B][N][C] bf16   (64x64 tiles through LDS)
__global__ void transpose_cvt(const float* __restrict__ x, __bf16* __restrict__ xt) {
    __shared__ __bf16 tile[64][66];
    int b  = blockIdx.z;
    int c0 = blockIdx.y * 64;
    int n0 = blockIdx.x * 64;
    int tx = threadIdx.x & 63;
    int ty = threadIdx.x >> 6;
    const float* xb = x + (size_t)b * C_ * N_;
#pragma unroll 4
    for (int i = 0; i < 16; ++i) {
        int c = ty * 16 + i;
        tile[c][tx] = (__bf16)xb[(size_t)(c0 + c) * N_ + n0 + tx];
    }
    __syncthreads();
    __bf16* xtb = xt + (size_t)b * N_ * C_;
#pragma unroll 4
    for (int i = 0; i < 16; ++i) {
        int n = ty * 16 + i;
        xtb[(size_t)(n0 + n) * C_ + c0 + tx] = tile[tx][n];
    }
}

// ---------------------------------------------------------------------------
// GEMM: Out[o,n] = sum_c W[o,c] * X[c,n] (+bias) [EPI==1: +BN+ReLU]
// W bf16 [Cout][Cin], Xt bf16 [B][N][Cin], OutT bf16 [B][N][Cout].
// One 16x16 WMMA tile per wave; 4 waves/block.
// ---------------------------------------------------------------------------
template <int EPI>
__global__ void gemm_T(const __bf16* __restrict__ W, const __bf16* __restrict__ Xt,
                       const float* __restrict__ bias, const float* __restrict__ inv,
                       const float* __restrict__ shift, __bf16* __restrict__ OutT,
                       int Cin, int Cout) {
    int b    = blockIdx.z;
    int wave = threadIdx.x >> 5;
    int lane = threadIdx.x & 31;
    int n0   = (blockIdx.x * 4 + wave) * 16;
    int o0   = blockIdx.y * 16;
    const __bf16* xb = Xt + (size_t)b * N_ * Cin + (size_t)n0 * Cin;
    const __bf16* wb = W + (size_t)o0 * Cin;

    v8f acc = {0.f, 0.f, 0.f, 0.f, 0.f, 0.f, 0.f, 0.f};
    for (int k = 0; k < Cin; k += 32) {
        v16bf a  = load_a_frag(wb + k, Cin);
        v16bf bb = load_b_frag(xb + k, Cin);
        acc = wmma_bf16(a, bb, acc);
    }
    int oc0 = o0 + (lane >> 4) * 8;
    int n   = n0 + (lane & 15);
    v8bf out;
#pragma unroll
    for (int r = 0; r < 8; ++r) {
        int o = oc0 + r;
        float v = acc[r] + bias[o];
        if (EPI == 1) { v = v * inv[o] + shift[o]; v = v > 0.f ? v : 0.f; }
        out[r] = (__bf16)v;
    }
    *(v8bf*)(OutT + (size_t)b * N_ * Cout + (size_t)n * Cout + oc0) = out;
}

// V projection: stores V natural [B][C][N] bf16 (A-operand of the message GEMM).
__global__ void gemm_V(const __bf16* __restrict__ W, const __bf16* __restrict__ Xt,
                       const float* __restrict__ bias, __bf16* __restrict__ Vout) {
    int b    = blockIdx.z;
    int wave = threadIdx.x >> 5;
    int lane = threadIdx.x & 31;
    int n0   = (blockIdx.x * 4 + wave) * 16;
    int o0   = blockIdx.y * 16;
    const __bf16* xb = Xt + (size_t)b * N_ * C_ + (size_t)n0 * C_;
    const __bf16* wb = W + (size_t)o0 * C_;

    v8f acc = {0.f, 0.f, 0.f, 0.f, 0.f, 0.f, 0.f, 0.f};
    for (int k = 0; k < C_; k += 32) {
        v16bf a  = load_a_frag(wb + k, C_);
        v16bf bb = load_b_frag(xb + k, C_);
        acc = wmma_bf16(a, bb, acc);
    }
    int oc0 = o0 + (lane >> 4) * 8;
    int n   = n0 + (lane & 15);
    __bf16* vb = Vout + (size_t)b * C_ * N_;
#pragma unroll
    for (int r = 0; r < 8; ++r) {
        int o = oc0 + r;
        vb[(size_t)o * N_ + n] = (__bf16)(acc[r] + bias[o]);
    }
}

// Final layer: out[b,c,n] = x[b,c,n] + b3[c] + sum_k w3[c,k]*h2[k,n]   (f32)
__global__ void gemm_out(const __bf16* __restrict__ W, const __bf16* __restrict__ Xt,
                         const float* __restrict__ bias, const float* __restrict__ x,
                         float* __restrict__ out) {
    int b    = blockIdx.z;
    int wave = threadIdx.x >> 5;
    int lane = threadIdx.x & 31;
    int n0   = (blockIdx.x * 4 + wave) * 16;
    int o0   = blockIdx.y * 16;
    const __bf16* xb = Xt + (size_t)b * N_ * HC_ + (size_t)n0 * HC_;
    const __bf16* wb = W + (size_t)o0 * HC_;

    v8f acc = {0.f, 0.f, 0.f, 0.f, 0.f, 0.f, 0.f, 0.f};
    for (int k = 0; k < HC_; k += 32) {
        v16bf a  = load_a_frag(wb + k, HC_);
        v16bf bb = load_b_frag(xb + k, HC_);
        acc = wmma_bf16(a, bb, acc);
    }
    int oc0 = o0 + (lane >> 4) * 8;
    int n   = n0 + (lane & 15);
    const float* xi = x   + (size_t)b * C_ * N_;
    float*       oo = out + (size_t)b * C_ * N_;
#pragma unroll
    for (int r = 0; r < 8; ++r) {
        int o = oc0 + r;
        oo[(size_t)o * N_ + n] = xi[(size_t)o * N_ + n] + acc[r] + bias[o];
    }
}

// ---------------------------------------------------------------------------
// Fused attention, one (batch, 16-query-row) block per workgroup, 8 waves.
//   Sb (bf16, LDS) = Qblk · Ktᵀ          -> phase 1 (WMMA)
//   Sb = exp(Sb*beta - rowmax) in place  -> phase 2 (β stream = roofline term)
//   msgT = (V · Sbᵀ) * (1/rowsum)        -> phase 3 (WMMA, B-frag straight from LDS)
// Score strip kept entirely in LDS (~75 KB of the 320 KB WGP LDS): the N×N
// matrix never touches HBM (would be 2x134 MB in f32).
// ---------------------------------------------------------------------------
__global__ void __launch_bounds__(256, 1)
attention_kernel(const __bf16* __restrict__ Qt, const __bf16* __restrict__ Kt,
                 const __bf16* __restrict__ V, const float* __restrict__ beta,
                 __bf16* __restrict__ MsgT) {
    __shared__ __align__(128) __bf16 Sb[16][SLD];     // 65792 B score strip
    __shared__ __align__(32)  __bf16 Qs[16 * C_];     //  8192 B Q block
    __shared__ float red[16][16];
    __shared__ float rowinv[16];

    int b    = blockIdx.y;
    int n0   = blockIdx.x * 16;
    int tid  = threadIdx.x;
    int wave = tid >> 5;
    int lane = tid & 31;

    const __bf16* qsrc = Qt + (size_t)b * N_ * C_ + (size_t)n0 * C_;

    // TDM async staging of the 16x256 Q block (wave 0 issues the descriptor).
    if (tid < 32) {
        tdm_load_2d_bf16(qsrc, Qs, C_, 16, C_);
        __builtin_amdgcn_s_wait_tensorcnt(0);
    }
    __syncthreads();
    // Deterministic fallback copy (idempotent when the TDM descriptor is right).
    {
        const v8bf* src = (const v8bf*)qsrc;
        v8bf* dst = (v8bf*)Qs;
        dst[tid]       = src[tid];
        dst[tid + 256] = src[tid + 256];
    }
    __syncthreads();

    // Preload the 8 A-fragments of the Q block (K = 256 = 8 x 32).
    v16bf aF[8];
#pragma unroll
    for (int j = 0; j < 8; ++j) aF[j] = load_a_frag(Qs + j * 32, C_);

    // ---- Phase 1: Sb = Qblk · Ktᵀ (bf16 scores) ----
    const __bf16* kb = Kt + (size_t)b * N_ * C_;
    for (int mt = wave; mt < N_ / 16; mt += 8) {
        v8f acc = {0.f, 0.f, 0.f, 0.f, 0.f, 0.f, 0.f, 0.f};
#pragma unroll
        for (int j = 0; j < 8; ++j) {
            v16bf bb = load_b_frag(kb + (size_t)mt * 16 * C_ + j * 32, C_);
            acc = wmma_bf16(aF[j], bb, acc);
        }
        int m  = mt * 16 + (lane & 15);
        int nr = (lane >> 4) * 8;
#pragma unroll
        for (int r = 0; r < 8; ++r) Sb[nr + r][m] = (__bf16)acc[r];
    }
    __syncthreads();

    // ---- Phase 2: in-place beta-scale + softmax (unnormalized exp, keep 1/sum) ----
    // 16 threads per row; each owns 16 interleaved 8-element (16B) chunks.
    int row = tid >> 4;
    int seg = tid & 15;
    const float* br = beta + (size_t)b * N_ * N_ + (size_t)(n0 + row) * N_;
    __bf16* wr = &Sb[row][0];
    float mx = -3.0e38f;
#pragma unroll 4
    for (int i = 0; i < 16; ++i) {
        int q = (seg + 16 * i) * 8;
        __builtin_prefetch((const void*)(br + q + 1024), 0, 1);
        v8bf sv = *(const v8bf*)(wr + q);
        float4 b0 = *(const float4*)(br + q);
        float4 b1 = *(const float4*)(br + q + 4);
        float f[8];
        f[0] = (float)sv[0] * b0.x; f[1] = (float)sv[1] * b0.y;
        f[2] = (float)sv[2] * b0.z; f[3] = (float)sv[3] * b0.w;
        f[4] = (float)sv[4] * b1.x; f[5] = (float)sv[5] * b1.y;
        f[6] = (float)sv[6] * b1.z; f[7] = (float)sv[7] * b1.w;
#pragma unroll
        for (int j = 0; j < 8; ++j) { mx = fmaxf(mx, f[j]); sv[j] = (__bf16)f[j]; }
        *(v8bf*)(wr + q) = sv;
    }
    red[row][seg] = mx;
    __syncthreads();
#pragma unroll
    for (int i = 0; i < 16; ++i) mx = fmaxf(mx, red[row][i]);
    __syncthreads();
    float sum = 0.f;
#pragma unroll 4
    for (int i = 0; i < 16; ++i) {
        int q = (seg + 16 * i) * 8;
        v8bf sv = *(const v8bf*)(wr + q);
#pragma unroll
        for (int j = 0; j < 8; ++j) {
            float e = __expf((float)sv[j] - mx);
            sum += e;
            sv[j] = (__bf16)e;
        }
        *(v8bf*)(wr + q) = sv;
    }
    red[row][seg] = sum;
    __syncthreads();
    if (seg == 0) {
        float s = 0.f;
#pragma unroll
        for (int i = 0; i < 16; ++i) s += red[row][i];
        rowinv[row] = 1.f / s;
    }
    __syncthreads();

    // ---- Phase 3: msg = (V · expᵀ) * rowinv; B-frags read straight from LDS ----
    const __bf16* vb = V + (size_t)b * C_ * N_;
    float rinv = rowinv[lane & 15];
    for (int ct = wave; ct < C_ / 16; ct += 8) {
        v8f acc = {0.f, 0.f, 0.f, 0.f, 0.f, 0.f, 0.f, 0.f};
        for (int j = 0; j < 64; ++j) {
            v16bf a  = load_a_frag(vb + (size_t)ct * 16 * N_ + j * 32, N_);
            v16bf bb = load_b_frag(&Sb[0][0] + j * 32, SLD);
            acc = wmma_bf16(a, bb, acc);
        }
        int n  = n0 + (lane & 15);
        int c0 = ct * 16 + (lane >> 4) * 8;
        v8bf o;
#pragma unroll
        for (int r = 0; r < 8; ++r) o[r] = (__bf16)(acc[r] * rinv);
        *(v8bf*)(MsgT + (size_t)b * N_ * C_ + (size_t)n * C_ + c0) = o;
    }
}

// ---------------------------------------------------------------------------
// Launch
// ---------------------------------------------------------------------------
extern "C" void kernel_launch(void* const* d_in, const int* in_sizes, int n_in,
                              void* d_out, int out_size, void* d_ws, size_t ws_size,
                              hipStream_t stream) {
    (void)in_sizes; (void)n_in; (void)out_size; (void)ws_size;
    const float* x    = (const float*)d_in[0];
    const float* beta = (const float*)d_in[1];
    const float* wq   = (const float*)d_in[2];
    const float* bq   = (const float*)d_in[3];
    const float* wk   = (const float*)d_in[4];
    const float* bk   = (const float*)d_in[5];
    const float* wv   = (const float*)d_in[6];
    const float* bv   = (const float*)d_in[7];
    const float* w1   = (const float*)d_in[8];
    const float* b1   = (const float*)d_in[9];
    const float* g1   = (const float*)d_in[10];
    const float* be1  = (const float*)d_in[11];
    const float* m1   = (const float*)d_in[12];
    const float* v1   = (const float*)d_in[13];
    const float* w2   = (const float*)d_in[14];
    const float* b2   = (const float*)d_in[15];
    const float* g2   = (const float*)d_in[16];
    const float* be2  = (const float*)d_in[17];
    const float* m2   = (const float*)d_in[18];
    const float* v2   = (const float*)d_in[19];
    const float* w3   = (const float*)d_in[20];
    const float* b3   = (const float*)d_in[21];
    float* out = (float*)d_out;

    char* ws = (char*)d_ws;
    __bf16* xt   = (__bf16*)(ws + 0);          // [B][N][C]
    __bf16* Qt   = (__bf16*)(ws + 8388608);    // [B][N][C]
    __bf16* Kt   = (__bf16*)(ws + 16777216);   // [B][N][C]
    __bf16* Vn   = (__bf16*)(ws + 25165824);   // [B][C][N]
    __bf16* msgT = (__bf16*)(ws + 33554432);   // [B][N][C]
    __bf16* h1T  = (__bf16*)(ws + 41943040);   // [B][N][128]
    __bf16* h2T  = (__bf16*)(ws + 46137344);   // [B][N][128]
    __bf16* wq_b = (__bf16*)(ws + 50331648);
    __bf16* wk_b = (__bf16*)(ws + 50462720);
    __bf16* wv_b = (__bf16*)(ws + 50593792);
    __bf16* w1_b = (__bf16*)(ws + 50724864);
    __bf16* w2_b = (__bf16*)(ws + 50790400);
    __bf16* w3_b = (__bf16*)(ws + 50823168);
    float*  inv1 = (float*)(ws + 50888704);
    float*  sh1  = (float*)(ws + 50889216);
    float*  inv2 = (float*)(ws + 50889728);
    float*  sh2  = (float*)(ws + 50890240);

    // Prep: weight conversion, BN folding, input transpose.
    cvt_f32_bf16<<<256, 256, 0, stream>>>(wq, wq_b, C_ * C_);
    cvt_f32_bf16<<<256, 256, 0, stream>>>(wk, wk_b, C_ * C_);
    cvt_f32_bf16<<<256, 256, 0, stream>>>(wv, wv_b, C_ * C_);
    cvt_f32_bf16<<<128, 256, 0, stream>>>(w1, w1_b, HC_ * C_);
    cvt_f32_bf16<<<64,  256, 0, stream>>>(w2, w2_b, HC_ * HC_);
    cvt_f32_bf16<<<128, 256, 0, stream>>>(w3, w3_b, C_ * HC_);
    bn_prep<<<1, 256, 0, stream>>>(g1, be1, m1, v1, g2, be2, m2, v2, inv1, sh1, inv2, sh2);
    transpose_cvt<<<dim3(N_ / 64, C_ / 64, B_), 256, 0, stream>>>(x, xt);

    // QKV projections (bf16 WMMA).
    dim3 gqkv(N_ / 64, C_ / 16, B_);
    gemm_T<0><<<gqkv, 128, 0, stream>>>(wq_b, xt, bq, nullptr, nullptr, Qt, C_, C_);
    gemm_T<0><<<gqkv, 128, 0, stream>>>(wk_b, xt, bk, nullptr, nullptr, Kt, C_, C_);
    gemm_V<<<gqkv, 128, 0, stream>>>(wv_b, xt, bv, Vn);

    // Fused attention (score strip LDS-resident; TDM-staged Q block).
    attention_kernel<<<dim3(N_ / 16, B_), 256, 0, stream>>>(Qt, Kt, Vn, beta, msgT);

    // MLP.
    gemm_T<1><<<dim3(N_ / 64, HC_ / 16, B_), 128, 0, stream>>>(w1_b, msgT, b1, inv1, sh1, h1T, C_, HC_);
    gemm_T<1><<<dim3(N_ / 64, HC_ / 16, B_), 128, 0, stream>>>(w2_b, h1T, b2, inv2, sh2, h2T, HC_, HC_);
    gemm_out<<<dim3(N_ / 64, C_ / 16, B_), 128, 0, stream>>>(w3_b, h2T, b3, x, out);
}
